// ScaledDotProductAttention_36077725287113
// MI455X (gfx1250) — compile-verified, matrix operand order checked by reading, not run on hardware
//
#include <hip/hip_runtime.h>

// ---------------------------------------------------------------------------
// Causal flash attention forward: B=2, H=16, S=2048, D=64 (fp32 in/out).
// gfx1250 (CDNA5): wave32, v_wmma_f32_16x16x32_f16 (f16 A/B, f32 accumulate).
//  - 64-key tiles: softmax bookkeeping (butterfly max, rescale, barriers)
//    amortized per key; 18 WMMAs per tile (8 QK^T + 2 row-sum + 8 PV)
//  - softmax in log2 domain: Q pre-scaled by log2(e)/sqrt(D); exponentials
//    are raw v_exp_f32 via __builtin_amdgcn_exp2f (no libm denorm guard --
//    sub-2^-126 probabilities flush to 0, which f16 P rounds to 0 anyway)
//  - row-sum of P via WMMA against an all-ones B matrix
//  - row-max via ds_swizzle xor-butterfly (1 DS op per step)
//  - key axis permuted (key t*16+ln -> slot 4*ln+t) consistently in P and V^T
//    so P stores as packed 4xf16 qwords and V^T stages as packed qwords
//  - next K/V tile prefetched (f16-converted) in registers: HBM overlaps WMMA
// ---------------------------------------------------------------------------

typedef __attribute__((ext_vector_type(16))) _Float16 v16h;
typedef __attribute__((ext_vector_type(8)))  _Float16 v8h;
typedef __attribute__((ext_vector_type(4)))  _Float16 v4h;
typedef __attribute__((ext_vector_type(8)))  float    v8f;

#define ATT_B 2
#define ATT_H 16
#define ATT_S 2048
#define ATT_D 64
#define BQ    64      // queries per block (4 waves x 16 rows)
#define KT    64      // keys per iteration (4 score tiles, 2 PV k-steps)
#define NW    4       // waves per block
#define PAD   8       // row padding (f16 elems): keep 16B align, spread banks

#define SHUF16 0,1,2,3,4,5,6,7,8,9,10,11,12,13,14,15

// raw v_exp_f32 (2^x), no denorm-range guard
#define EXP2 __builtin_amdgcn_exp2f

// xor-butterfly lane swap within 16-lane group: ds_swizzle group-of-32 mode,
// imm = {xor_mask[4:0]<<10 | or=0<<5 | and=0x1f}
template <int IMM>
__device__ __forceinline__ float ds_swz(float x) {
    return __int_as_float(__builtin_amdgcn_ds_swizzle(__float_as_int(x), IMM));
}
__device__ __forceinline__ float grp16_max(float x) {
    x = fmaxf(x, ds_swz<0x041F>(x));   // xor 1
    x = fmaxf(x, ds_swz<0x081F>(x));   // xor 2
    x = fmaxf(x, ds_swz<0x101F>(x));   // xor 4
    x = fmaxf(x, ds_swz<0x201F>(x));   // xor 8
    return x;
}

__global__ __launch_bounds__(128)
void fa_causal_fwd_gfx1250(const float* __restrict__ Q,
                           const float* __restrict__ K,
                           const float* __restrict__ V,
                           float* __restrict__ O)
{
    __shared__ __align__(16) _Float16 lds_k [KT][ATT_D + PAD];    // [key][d]
    __shared__ __align__(16) _Float16 lds_vt[ATT_D][KT + PAD];    // [d][perm key]
    __shared__ __align__(16) _Float16 lds_p [NW][16][KT + PAD];   // [wave][row][perm key]

    const int tid  = threadIdx.x;
    const int wave = tid >> 5;
    const int lane = tid & 31;
    const int ln   = lane & 15;   // column / M index within 16-lane group
    const int g    = lane >> 4;   // half-wave selector (K-chunk / row+8 group)

    const int blocks_per_head = ATT_S / BQ;
    const int bh   = blockIdx.x / blocks_per_head;
    const int qblk = blockIdx.x % blocks_per_head;

    const size_t head_off = (size_t)bh * ATT_S * ATT_D;
    const float* gQ = Q + head_off;
    const float* gK = K + head_off;
    const float* gV = V + head_off;
    float*       gO = O + head_off;

    const int   qbase = qblk * BQ + wave * 16;
    // log2-domain softmax: scores come out pre-multiplied by log2(e)/sqrt(D),
    // so exp(s-m) == exp2(s'-m') and every exponential is a bare v_exp_f32.
    const float scale = 0.125f * 1.44269504088896340736f;

    // ---- Q fragments: A-matrix 16x32 layout (ISA 7.12.2), pre-scaled ------
    v16h qa[2];
    {
        const float* qp = gQ + (size_t)(qbase + ln) * ATT_D;
        #pragma unroll
        for (int c = 0; c < 2; ++c) {
            const float4 a0 = *(const float4*)(qp + c * 32 + g * 8);
            const float4 a1 = *(const float4*)(qp + c * 32 + g * 8 + 4);
            const float4 b0 = *(const float4*)(qp + c * 32 + 16 + g * 8);
            const float4 b1 = *(const float4*)(qp + c * 32 + 16 + g * 8 + 4);
            v16h f;
            f[0]  = (_Float16)(a0.x * scale); f[1]  = (_Float16)(a0.y * scale);
            f[2]  = (_Float16)(a0.z * scale); f[3]  = (_Float16)(a0.w * scale);
            f[4]  = (_Float16)(a1.x * scale); f[5]  = (_Float16)(a1.y * scale);
            f[6]  = (_Float16)(a1.z * scale); f[7]  = (_Float16)(a1.w * scale);
            f[8]  = (_Float16)(b0.x * scale); f[9]  = (_Float16)(b0.y * scale);
            f[10] = (_Float16)(b0.z * scale); f[11] = (_Float16)(b0.w * scale);
            f[12] = (_Float16)(b1.x * scale); f[13] = (_Float16)(b1.y * scale);
            f[14] = (_Float16)(b1.z * scale); f[15] = (_Float16)(b1.w * scale);
            qa[c] = f;
        }
    }

    // all-ones B fragment: row-sum of P via WMMA (D[m][n] = sum_k P[m][k])
    v16h ones;
    #pragma unroll
    for (int j = 0; j < 16; ++j) ones[j] = (_Float16)1.0f;

    v8f o_acc[4];
    float m8[8], l8[8];
    #pragma unroll
    for (int t = 0; t < 4; ++t) o_acc[t] = (v8f){};
    #pragma unroll
    for (int r = 0; r < 8; ++r) { m8[r] = -1.0e30f; l8[r] = 0.0f; }

    const int ntiles = qblk + 1;   // 64-key tiles covering keys 0..qblk*64+63

    // ---- staging registers (f16-converted at prefetch) --------------------
    // K: thread holds 8x (row, 4 consecutive d)     -> b64 stores, row-major
    // V: thread holds 8x (d, keys m,m+16,m+32,m+48) -> b64 stores, perm slots
    v4h kreg[8], vreg[8];

    #define PREFETCH(KB)                                                      \
        {                                                                     \
            const float* kp = gK + (size_t)(KB) * ATT_D;                      \
            const float* vp = gV + (size_t)(KB) * ATT_D;                      \
            _Pragma("unroll")                                                 \
            for (int i = 0; i < 8; ++i) {                                     \
                const int idx  = tid + i * 128;        /* 0..1023 */          \
                const int krow = idx >> 4;                                    \
                const int kc4  = (idx & 15) << 2;                             \
                const float4 kv = *(const float4*)(kp + krow * ATT_D + kc4);  \
                kreg[i][0] = (_Float16)kv.x; kreg[i][1] = (_Float16)kv.y;     \
                kreg[i][2] = (_Float16)kv.z; kreg[i][3] = (_Float16)kv.w;     \
                const int m = idx >> 6;                                       \
                const int d = idx & 63;                                       \
                vreg[i][0] = (_Float16)vp[(m     ) * ATT_D + d];              \
                vreg[i][1] = (_Float16)vp[(m + 16) * ATT_D + d];              \
                vreg[i][2] = (_Float16)vp[(m + 32) * ATT_D + d];              \
                vreg[i][3] = (_Float16)vp[(m + 48) * ATT_D + d];              \
            }                                                                 \
        }

    PREFETCH(0)

    for (int kt = 0; kt < ntiles; ++kt) {
        const int kbase = kt * KT;

        __syncthreads();   // previous iteration's LDS reads complete

        // ---- stage current tile from regs ---------------------------------
        #pragma unroll
        for (int i = 0; i < 8; ++i) {
            const int idx  = tid + i * 128;
            const int krow = idx >> 4;
            const int kc4  = (idx & 15) << 2;
            *(v4h*)&lds_k[krow][kc4] = kreg[i];
            const int m = idx >> 6;
            const int d = idx & 63;
            *(v4h*)&lds_vt[d][4 * m] = vreg[i];   // keys m,m+16,m+32,m+48
        }
        __syncthreads();

        // ---- prefetch next tile (overlaps with WMMA work below) -----------
        if (kt + 1 < ntiles) PREFETCH(kbase + KT)

        // ---- scores S' = (Q*log2e/sqrt(D)) * K^T : four 16x16 tiles -------
        v8f s[4];
        #pragma unroll
        for (int t = 0; t < 4; ++t) s[t] = (v8f){};
        #pragma unroll
        for (int c = 0; c < 2; ++c) {
            const int off = c * 32 + g * 8;
            #pragma unroll
            for (int t = 0; t < 4; ++t) {      // keys kbase + t*16 + 0..15
                const v8h lo = *(const v8h*)&lds_k[t * 16 + ln][off];
                const v8h hi = *(const v8h*)&lds_k[t * 16 + ln][off + 16];
                const v16h b = __builtin_shufflevector(lo, hi, SHUF16);
                s[t] = __builtin_amdgcn_wmma_f32_16x16x32_f16(
                           false, qa[c], false, b, (short)0, s[t], false, false);
            }
        }

        // ---- online softmax (log2 domain): max via swizzle butterfly ------
        const bool need_mask = (kbase + KT - 1) > qbase;   // diagonal tile only
        #pragma unroll
        for (int r = 0; r < 8; ++r) {
            float sv[4];
            #pragma unroll
            for (int t = 0; t < 4; ++t) sv[t] = s[t][r];
            if (need_mask) {
                const int qrow = qbase + r + 8 * g;
                #pragma unroll
                for (int t = 0; t < 4; ++t)
                    if (kbase + t * 16 + ln > qrow) sv[t] = -1.0e9f;
            }
            const float rmax = grp16_max(fmaxf(fmaxf(sv[0], sv[1]),
                                               fmaxf(sv[2], sv[3])));
            const float mnew = fmaxf(m8[r], rmax);
            const float fs   = EXP2(m8[r] - mnew);   // bare v_exp_f32
            m8[r] = mnew;
            v4h pp;   // key kbase + t*16 + ln -> perm slot 4*ln + t
            #pragma unroll
            for (int t = 0; t < 4; ++t) pp[t] = (_Float16)EXP2(sv[t] - mnew);
            *(v4h*)&lds_p[wave][r + 8 * g][4 * ln] = pp;
            l8[r] *= fs;
            #pragma unroll
            for (int t = 0; t < 4; ++t) o_acc[t][r] *= fs;
        }

        // same-wave LDS RAW: drain DS counter before A-layout reads
        asm volatile("s_wait_dscnt 0" ::: "memory");

        // ---- P as two A-fragments (16x64, permuted key order) -------------
        v16h pa[2];
        #pragma unroll
        for (int c = 0; c < 2; ++c) {
            const v8h plo = *(const v8h*)&lds_p[wave][ln][c * 32 + g * 8];
            const v8h phi = *(const v8h*)&lds_p[wave][ln][c * 32 + 16 + g * 8];
            pa[c] = __builtin_shufflevector(plo, phi, SHUF16);
        }

        // row-sum of P via WMMA: ls[r] = sum_k P[row r+8g][k]
        v8f ls = (v8f){};
        ls = __builtin_amdgcn_wmma_f32_16x16x32_f16(
                 false, pa[0], false, ones, (short)0, ls, false, false);
        ls = __builtin_amdgcn_wmma_f32_16x16x32_f16(
                 false, pa[1], false, ones, (short)0, ls, false, false);
        #pragma unroll
        for (int r = 0; r < 8; ++r) l8[r] += ls[r];

        // ---- O += P * V (V^T columns as B-fragments, same key perm) -------
        #pragma unroll
        for (int c = 0; c < 2; ++c) {
            const int off = c * 32 + g * 8;
            #pragma unroll
            for (int t = 0; t < 4; ++t) {
                const v8h vlo = *(const v8h*)&lds_vt[t * 16 + ln][off];
                const v8h vhi = *(const v8h*)&lds_vt[t * 16 + ln][off + 16];
                const v16h vb = __builtin_shufflevector(vlo, vhi, SHUF16);
                o_acc[t] = __builtin_amdgcn_wmma_f32_16x16x32_f16(
                               false, pa[c], false, vb, (short)0, o_acc[t],
                               false, false);
            }
        }
    }

    // ---- epilogue: O = acc / l, fp32 out ----------------------------------
    #pragma unroll
    for (int r = 0; r < 8; ++r) {
        const float inv = 1.0f / l8[r];
        const size_t rowoff = (size_t)(qbase + r + 8 * g) * ATT_D;
        #pragma unroll
        for (int t = 0; t < 4; ++t)
            gO[rowoff + t * 16 + ln] = o_acc[t][r] * inv;
    }
}

extern "C" void kernel_launch(void* const* d_in, const int* in_sizes, int n_in,
                              void* d_out, int out_size, void* d_ws, size_t ws_size,
                              hipStream_t stream) {
    (void)in_sizes; (void)n_in; (void)d_ws; (void)ws_size; (void)out_size;
    const float* q = (const float*)d_in[0];
    const float* k = (const float*)d_in[1];
    const float* v = (const float*)d_in[2];
    // d_in[3] is the causal mask; causality is computed analytically.
    float* out = (float*)d_out;

    dim3 grid(ATT_B * ATT_H * (ATT_S / BQ));
    dim3 block(NW * 32);
    fa_causal_fwd_gfx1250<<<grid, block, 0, stream>>>(q, k, v, out);
}
